// SpatioConvLayer_86552180949318
// MI455X (gfx1250) — matrix operand order repaired and checked by reading, not compiled, not run on hardware
//
#include <hip/hip_runtime.h>
#include <stdint.h>

typedef __attribute__((ext_vector_type(16))) _Float16 v16h;
typedef __attribute__((ext_vector_type(8)))  _Float16 v8h;
typedef __attribute__((ext_vector_type(4)))  _Float16 v4h;
typedef __attribute__((ext_vector_type(8)))  float    v8f;
typedef __attribute__((ext_vector_type(4)))  int      v4i;

#define KC    64            // reduction chunk staged in LDS
#define KCP   72            // padded pitch (halves): 144B = 36 dwords -> conflict-free
#define MT    128           // m-tile per workgroup
#define NSTEP 48            // 3 k-taps x 16 chunks

// ---- async global->LDS (CDNA5 path), with compile-probed fallback ----------
#if __has_builtin(__builtin_amdgcn_global_load_async_to_lds_b128)
#define HAVE_ASYNC 1
typedef __attribute__((address_space(1))) v4i gv4i;   // global int4 payload
typedef __attribute__((address_space(3))) v4i lv4i;   // LDS int4 payload
#else
#define HAVE_ASYNC 0
#endif

__device__ __forceinline__ void cp16_async(const _Float16* g, _Float16* s) {
#if HAVE_ASYNC
  __builtin_amdgcn_global_load_async_to_lds_b128(
      (gv4i*)(uintptr_t)g, (lv4i*)(uint32_t)(uintptr_t)s, 0, 0);
#else
  *(v8h*)s = *(const v8h*)g;     // sync fallback: b128 load + ds_store
#endif
}

__device__ __forceinline__ void wait_async0() {
#if HAVE_ASYNC
#if __has_builtin(__builtin_amdgcn_s_wait_asynccnt)
  __builtin_amdgcn_s_wait_asynccnt(0);
#else
  asm volatile("s_wait_asynccnt 0x0" ::: "memory");
#endif
#endif
}

// Build a 16-half WMMA fragment from two contiguous 16B runs in LDS/global.
__device__ __forceinline__ v16h ldfrag(const _Float16* lo, const _Float16* hi) {
  v8h a = *(const v8h*)lo;
  v8h b = *(const v8h*)hi;
  return __builtin_shufflevector(a, b, 0,1,2,3,4,5,6,7,8,9,10,11,12,13,14,15);
}

// ---------------- prologue kernels (precision conversion / layout) ----------

__global__ __launch_bounds__(256) void cvt_x_f16(const float* __restrict__ in,
                                                 _Float16* __restrict__ outh, int n4) {
  int i = blockIdx.x * 256 + threadIdx.x;
  if (i < n4) {
    float4 f = ((const float4*)in)[i];
    v4h h4 = { (_Float16)f.x, (_Float16)f.y, (_Float16)f.z, (_Float16)f.w };
    ((v4h*)outh)[i] = h4;
  }
}

// kernel[n=1024][mk=3072] (f32) -> KhT[mk=3072][n=1024] (f16), tiled transpose
__global__ __launch_bounds__(256) void cvt_transpose_kernel(const float* __restrict__ kin,
                                                            _Float16* __restrict__ kT) {
  __shared__ float tile[32][33];
  const int tx = threadIdx.x, ty = threadIdx.y;           // block (32, 8)
  const int mk0 = blockIdx.x * 32, n0 = blockIdx.y * 32;
#pragma unroll
  for (int i = 0; i < 32; i += 8)
    tile[ty + i][tx] = kin[(long)(n0 + ty + i) * 3072 + mk0 + tx];
  __syncthreads();
#pragma unroll
  for (int i = 0; i < 32; i += 8)
    kT[(long)(mk0 + ty + i) * 1024 + n0 + tx] = (_Float16)tile[tx][ty + i];
}

// theta[(c*3+k)][o] (f32) -> thT[k][o][c] (f16)
__global__ __launch_bounds__(256) void cvt_theta(const float* __restrict__ th,
                                                 _Float16* __restrict__ thT) {
  int i = blockIdx.x * 256 + threadIdx.x;
  if (i < 3 * 64 * 64) {
    int k = i >> 12;
    int o = (i >> 6) & 63;
    int c = i & 63;
    thT[i] = (_Float16)th[(c * 3 + k) * 64 + o];
  }
}

// ---------------- fused main kernel ----------------------------------------
// One workgroup per (b, t, m-tile).  gc[64, MT] = sum_k thT_k * (Xh_bt * Kr_k)
// epilogue: out = relu(gc + bias + x)   (c_in == c_out identity residual)

// Issue async staging of linear step s (= k*16 + chunk) into given buffers.
__device__ __forceinline__ void stage_chunk(int s, int m0, long xrow0,
                                            const _Float16* __restrict__ xh,
                                            const _Float16* __restrict__ khT,
                                            _Float16* XAb, _Float16* BTb,
                                            int xr, int xs, int br, int bs) {
  const int k  = s >> 4;
  const int n0 = (s & 15) * KC;
  // XA: X_bt[c=xr][n0+xs .. +15]  (64 rows x 128B per chunk)
  const _Float16* gx = xh + xrow0 + (long)xr * 12288 + n0 + xs;
  _Float16* sx = XAb + xr * KCP + xs;
  cp16_async(gx,     sx);
  cp16_async(gx + 8, sx + 8);
  // BT: KhT[k][m0+br][n0+bs .. +31]  (128 rows x 128B per chunk)
  const _Float16* gb = khT + (((long)k * 1024 + m0) + br) * 1024 + n0 + bs;
  _Float16* sb = BTb + br * KCP + bs;
#pragma unroll
  for (int j = 0; j < 4; ++j) cp16_async(gb + j * 8, sb + j * 8);
}

__global__ __launch_bounds__(256) void stgcn_main(
    const _Float16* __restrict__ xh,    // [B,64,T,1024] f16
    const _Float16* __restrict__ khT,   // [3,1024(m),1024(n)] f16
    const _Float16* __restrict__ thT,   // [3,64(o),64(c)] f16
    const float*    __restrict__ bias,  // [64]
    const float*    __restrict__ xres,  // [B,64,T,1024] f32 (residual)
    float*          __restrict__ out)   // [B,64,T,1024] f32
{
  __shared__ __align__(16) _Float16 XA2[2][64 * KCP];   // 2 x 9 KB
  __shared__ __align__(16) _Float16 BT2[2][MT * KCP];   // 2 x 18 KB (epilogue reuses)

  const int tid  = threadIdx.x;
  const int lane = tid & 31;
  const int wave = tid >> 5;
  const int h    = lane >> 4;      // half-wave
  const int ln   = lane & 15;
  const int m0   = blockIdx.x * MT;
  const int t    = blockIdx.y;
  const int b    = blockIdx.z;
  const int mw   = wave * 16;      // this wave's 16 output columns inside the tile

  const long xrow0 = ((long)b * 768 + t) * 1024;    // + c*12288 + n
  const int xr = tid >> 2, xs = (tid & 3) * 16;     // XA staging: 64 rows x 32B/thread
  const int br = tid >> 1, bs = (tid & 1) * 32;     // BT staging: 128 rows x 64B/thread

  const v8f zero8 = {0.f,0.f,0.f,0.f,0.f,0.f,0.f,0.f};
  v8f gc[4] = {zero8, zero8, zero8, zero8};         // 4 o-tiles x 16 cols (persist)

  // prologue: stage step 0 into buffer 0
  stage_chunk(0, m0, xrow0, xh, khT, XA2[0], BT2[0], xr, xs, br, bs);

  for (int k = 0; k < 3; ++k) {
    v8f w[4] = {zero8, zero8, zero8, zero8};        // 4 c-tiles x 16 cols (per k)

#pragma unroll 2
    for (int ch = 0; ch < 16; ++ch) {
      const int s   = k * 16 + ch;
      const int cur = ch & 1;                       // k*16 is even
      wait_async0();             // this wave's async transfers for chunk s landed
      __syncthreads();           // => every wave's chunk-s data visible in LDS
      if (s + 1 < NSTEP)         // overlap next chunk's loads with this compute
        stage_chunk(s + 1, m0, xrow0, xh, khT, XA2[1 - cur], BT2[1 - cur],
                    xr, xs, br, bs);

      const _Float16* XA = XA2[cur];
      const _Float16* BT = BT2[cur];
#pragma unroll
      for (int kk = 0; kk < KC; kk += 32) {
        // B fragment: B[K=n][N=m], lanes = m columns, K contiguous in LDS row
        const _Float16* bp = &BT[(mw + ln) * KCP + kk + h * 8];
        v16h bf = ldfrag(bp, bp + 16);
#pragma unroll
        for (int ct = 0; ct < 4; ++ct) {
          // A fragment: A[M=c][K=n]; elems 0-7: K=kk+8h+i, 8-15: K=kk+16+8h+i
          const _Float16* ap = &XA[(ct * 16 + ln) * KCP + kk + h * 8];
          v16h af = ldfrag(ap, ap + 16);
          w[ct] = __builtin_amdgcn_wmma_f32_16x16x32_f16(
              false, af, false, bf, (short)0, w[ct], false, false);
        }
      }
    }

    // ---- epilogue for this k: gc[o,m] += sum_c th_k[c,o] * W_k[c,m] ------
    // Last chunk index is 15 (odd) -> it lived in buffer 1; reuse it as the
    // W restage buffer (each wave touches only its own 16 rows; the next
    // write into buffer 1 happens after a later barrier).
    _Float16* WT = BT2[1];
#pragma unroll
    for (int ct = 0; ct < 4; ++ct) {
      v8h hw;
#pragma unroll
      for (int v = 0; v < 8; ++v) hw[v] = (_Float16)w[ct][v];
      // D elem (lane, v) is W[c = ct*16 + 8h + v][m' = mw+ln]
      *(v8h*)(&WT[(mw + ln) * KCP + ct * 16 + h * 8]) = hw;
    }
#pragma unroll
    for (int ot = 0; ot < 4; ++ot) {
#pragma unroll
      for (int cc = 0; cc < 64; cc += 32) {
        const _Float16* tp = thT + ((long)k * 64 + ot * 16 + ln) * 64 + cc + h * 8;
        v16h af = ldfrag(tp, tp + 16);
        const _Float16* wp = &WT[(mw + ln) * KCP + cc + h * 8];
        v16h bf = ldfrag(wp, wp + 16);
        gc[ot] = __builtin_amdgcn_wmma_f32_16x16x32_f16(
            false, af, false, bf, (short)0, gc[ot], false, false);
      }
    }
  }

  // ---- bias + identity residual + relu + store -----------------------------
  const int m = m0 + mw + ln;
#pragma unroll
  for (int ot = 0; ot < 4; ++ot) {
#pragma unroll
    for (int v = 0; v < 8; ++v) {
      const int o = ot * 16 + 8 * h + v;
      const long idx = ((long)b * 64 + o) * 12288 + (long)t * 1024 + m;
      float val = gc[ot][v] + bias[o] + xres[idx];
      out[idx] = val > 0.f ? val : 0.f;
    }
  }
}

// ---------------- host launch ----------------------------------------------

extern "C" void kernel_launch(void* const* d_in, const int* in_sizes, int n_in,
                              void* d_out, int out_size, void* d_ws, size_t ws_size,
                              hipStream_t stream) {
  (void)in_sizes; (void)n_in; (void)out_size; (void)ws_size;
  const float* x     = (const float*)d_in[0];   // [16,64,12,1024]
  const float* kern  = (const float*)d_in[1];   // [1024,3072]
  const float* theta = (const float*)d_in[2];   // [192,64]
  const float* bias  = (const float*)d_in[3];   // [64]

  _Float16* xh  = (_Float16*)d_ws;              // 12,582,912 halves (25.2 MB)
  _Float16* khT = xh + 12582912;                // 3,145,728 halves  (6.3 MB)
  _Float16* thT = khT + 3145728;                // 12,288 halves

  cvt_x_f16<<<dim3(12288), dim3(256), 0, stream>>>(x, xh, 12582912 / 4);
  cvt_transpose_kernel<<<dim3(96, 32), dim3(32, 8), 0, stream>>>(kern, khT);
  cvt_theta<<<dim3(48), dim3(256), 0, stream>>>(theta, thT);
  stgcn_main<<<dim3(8, 12, 16), dim3(256), 0, stream>>>(xh, khT, thT, bias, x,
                                                        (float*)d_out);
}
